// TensorEmbedding_65060164599843
// MI455X (gfx1250) — compile-verified, matrix operand order checked by reading, not run on hardware
//
#include <hip/hip_runtime.h>
#include <hip/hip_bf16.h>
#include <math.h>

// ---------------------------------------------------------------------------
// Problem constants (match reference)
// ---------------------------------------------------------------------------
#define N_NODES 5000
#define N_EDGES 80000
#define HIDDEN  128
#define NUM_RBF 64
#define CUTOFF  5.0f
#define LN_EPS  1e-5f

// LDS-staged weight block: W_emb2 [128,256] | Wd1 | Wd2 | Wd3 [128,64] each
#define LDS_W_HALFS (HIDDEN * 256 + 3 * HIDDEN * NUM_RBF)   // 57344 halves
#define LDS_W_BYTES (LDS_W_HALFS * 2)                       // 114688 B (<320KB WGP LDS)

typedef __attribute__((ext_vector_type(16))) _Float16 v16h;
typedef __attribute__((ext_vector_type(8)))  float    v8f;

static __device__ __forceinline__ v8f wmma_f16(v16h a, v16h b, v8f c) {
    // D = A(16x32 f16) * B(32x16 f16) + C(16x16 f32)
    return __builtin_amdgcn_wmma_f32_16x16x32_f16(
        /*neg_a=*/false, a, /*neg_b=*/false, b,
        /*c_mod=*/(short)0, c, /*reuse_a=*/false, /*reuse_b=*/false);
}

// 16-bit A/B fragment K-index for element j of the per-lane v16h
// (lanes 0-15: K in {0..7,16..23}; lanes 16-31: K in {8..15,24..31})
static __device__ __forceinline__ int kmap(int lane, int j) {
    return ((lane & 16) ? 8 : 0) + (j & 7) + ((j >> 3) << 4);
}

// ---------------------------------------------------------------------------
// 0a) zero helper
// ---------------------------------------------------------------------------
__global__ void zero_f32_kernel(float* __restrict__ p, int n) {
    for (int i = blockIdx.x * blockDim.x + threadIdx.x; i < n;
         i += gridDim.x * blockDim.x)
        p[i] = 0.0f;
}

// 0b) f32 -> f16 staging (weights are stationary: convert once per launch)
__global__ void cvt_f16_kernel(const float* __restrict__ src,
                               _Float16* __restrict__ dst, int n) {
    for (int i = blockIdx.x * blockDim.x + threadIdx.x; i < n;
         i += gridDim.x * blockDim.x)
        dst[i] = (_Float16)src[i];
}

// ---------------------------------------------------------------------------
// 1) Fused edge kernel: Zij GEMM (K=256), f1/f2/f3 GEMMs (K=64), cutoff,
//    rank-structured scatter-add into acc[node][h][10].
//    One wave per 16-edge tile, 8 waves / block, weights staged in LDS and
//    shared block-wide. Channel chunks processed in two halves of 4 so the
//    live accumulator set (32 VGPRs) plus the F-phase working set never
//    exceeds the register budget -> zero scratch traffic.
// ---------------------------------------------------------------------------
__global__ void __launch_bounds__(256, 1)
edge_kernel(const int*      __restrict__ z,
            const int*      __restrict__ edge_index, // [2,E]
            const float*    __restrict__ edge_weight,
            const float*    __restrict__ edge_vec,   // [E,3]
            const float*    __restrict__ edge_attr,  // [E,64] f32
            const _Float16* __restrict__ emb16,      // [128,128]
            const _Float16* __restrict__ wblk16,     // Wemb2|Wd1|Wd2|Wd3 f16
            const float*    __restrict__ b_emb2,     // [128]
            const float*    __restrict__ bd1,
            const float*    __restrict__ bd2,
            const float*    __restrict__ bd3,
            float* __restrict__ acc)                 // [N,128,10]
{
    extern __shared__ _Float16 lds[];
    // cooperative block-wide weight staging: 57344 halves = 7168 x uint4
    {
        const uint4* src = (const uint4*)wblk16;
        uint4* dst = (uint4*)lds;
        for (int i = threadIdx.x; i < LDS_W_HALFS / 8; i += 256)
            dst[i] = src[i];
    }
    __syncthreads();
    const _Float16* sW   = lds;                        // [128][256]
    const _Float16* sWd1 = lds + HIDDEN * 256;         // [128][64]
    const _Float16* sWd2 = sWd1 + HIDDEN * NUM_RBF;
    const _Float16* sWd3 = sWd2 + HIDDEN * NUM_RBF;

    const int lane = threadIdx.x & 31;
    const int wave = threadIdx.x >> 5;
    const int tile = blockIdx.x * 8 + wave;
    if (tile * 16 >= N_EDGES) return;           // wave-uniform (never hit: 5000 tiles)
    const int e0 = tile * 16;

    const int mA = lane & 15;                    // A-fragment row
    const int eA = e0 + mA;                      // E is a multiple of 16
    const int ch = lane & 15;                    // output-channel within chunk

    // prefetch next tile's edge_attr row (speculative, L2)
    {
        const int eN = eA + 16 * 8;
        if (eN < N_EDGES)
            __builtin_prefetch(edge_attr + (size_t)eN * NUM_RBF, 0, 0);
    }

    const int zs = z[edge_index[eA]];
    const int zd = z[edge_index[N_EDGES + eA]];
    const _Float16* embS = emb16 + (size_t)zs * HIDDEN;
    const _Float16* embD = emb16 + (size_t)zd * HIDDEN;

    // ---------------- edge_attr A fragments (K=64 -> 2 steps) ---------------
    v16h Af[2];
#pragma unroll
    for (int t = 0; t < 2; ++t) {
        const int k0 = t * 32;
#pragma unroll
        for (int j = 0; j < 16; ++j)
            Af[t][j] = (_Float16)edge_attr[(size_t)eA * NUM_RBF + k0 + kmap(lane, j)];
    }

    // ---------------- epilogue edge data: rows m = (lane>>4)*8 + r ----------
    const int rbase = (lane >> 4) << 3;
    int   er_node[8];
    float er_cut[8], er_vx[8], er_vy[8], er_vz[8];
#pragma unroll
    for (int r = 0; r < 8; ++r) {
        const int ee = e0 + rbase + r;
        er_node[r]  = edge_index[ee];            // src
        const float w = edge_weight[ee];
        const float c = 0.5f * (cosf(w * (3.14159265358979f / CUTOFF)) + 1.0f);
        er_cut[r]   = (w < CUTOFF) ? c : 0.0f;
        er_vx[r] = edge_vec[(size_t)ee * 3 + 0];
        er_vy[r] = edge_vec[(size_t)ee * 3 + 1];
        er_vz[r] = edge_vec[(size_t)ee * 3 + 2];
    }

    // =============== two halves of 4 channel-chunks each ====================
#pragma unroll 1
    for (int half = 0; half < 2; ++half) {
        const int ncb = half * 4;                // first chunk of this half

        // ---------------- Z phase: K-outer (rolled), N-inner ----------------
        v8f zaccs[4];
#pragma unroll
        for (int i = 0; i < 4; ++i) zaccs[i] = (v8f){};

        // K-steps 0..3: features from Z[src] (K < 128)
#pragma unroll 1
        for (int s = 0; s < 4; ++s) {
            const int k0 = s * 32;
            v16h Zf;
#pragma unroll
            for (int j = 0; j < 16; ++j)
                Zf[j] = embS[k0 + kmap(lane, j)];
#pragma unroll
            for (int i = 0; i < 4; ++i) {
                const int nn = (ncb + i) * 16 + ch;
                v16h B;
#pragma unroll
                for (int j = 0; j < 16; ++j)
                    B[j] = sW[nn * 256 + k0 + kmap(lane, j)];
                zaccs[i] = wmma_f16(Zf, B, zaccs[i]);
            }
        }
        // K-steps 4..7: features from Z[dst] (K >= 128)
#pragma unroll 1
        for (int s = 4; s < 8; ++s) {
            const int k0 = s * 32;
            v16h Zf;
#pragma unroll
            for (int j = 0; j < 16; ++j)
                Zf[j] = embD[k0 - HIDDEN + kmap(lane, j)];
#pragma unroll
            for (int i = 0; i < 4; ++i) {
                const int nn = (ncb + i) * 16 + ch;
                v16h B;
#pragma unroll
                for (int j = 0; j < 16; ++j)
                    B[j] = sW[nn * 256 + k0 + kmap(lane, j)];
                zaccs[i] = wmma_f16(Zf, B, zaccs[i]);
            }
        }

        // ---------------- F phase + scatter, per 16-channel chunk -----------
#pragma unroll 1
        for (int i = 0; i < 4; ++i) {
            const int nn = (ncb + i) * 16 + ch;

            v8f g1 = {}, g2 = {}, g3 = {};
#pragma unroll
            for (int t = 0; t < 2; ++t) {
                const int k0 = t * 32;
                v16h B1, B2, B3;
#pragma unroll
                for (int j = 0; j < 16; ++j) {
                    const int k = k0 + kmap(lane, j);
                    B1[j] = sWd1[nn * NUM_RBF + k];
                    B2[j] = sWd2[nn * NUM_RBF + k];
                    B3[j] = sWd3[nn * NUM_RBF + k];
                }
                g1 = wmma_f16(Af[t], B1, g1);
                g2 = wmma_f16(Af[t], B2, g2);
                g3 = wmma_f16(Af[t], B3, g3);
            }

            const float be = b_emb2[nn];
            const float b1 = bd1[nn], b2 = bd2[nn], b3 = bd3[nn];
            const v8f zc = zaccs[i];
#pragma unroll
            for (int r = 0; r < 8; ++r) {
                const float C  = er_cut[r] * (zc[r] + be);
                const float f1 = (g1[r] + b1) * C;
                const float f2 = (g2[r] + b2) * C;
                const float f3 = (g3[r] + b3) * C;
                const float vx = er_vx[r], vy = er_vy[r], vz = er_vz[r];
                float* p = acc + ((size_t)er_node[r] * HIDDEN + nn) * 10;
                atomicAdd(p + 0, f1);                       // I (diag scalar)
                atomicAdd(p + 1, f2 * vx);                  // A vector
                atomicAdd(p + 2, f2 * vy);
                atomicAdd(p + 3, f2 * vz);
                atomicAdd(p + 4, f3 * (vx * vx - (1.0f / 3.0f)));   // S traceless
                atomicAdd(p + 5, f3 * (vy * vy - (1.0f / 3.0f)));
                atomicAdd(p + 6, f3 * (vz * vz - (1.0f / 3.0f)));
                atomicAdd(p + 7, f3 * vx * vy);
                atomicAdd(p + 8, f3 * vx * vz);
                atomicAdd(p + 9, f3 * vy * vz);
            }
        }
    }
}

// ---------------------------------------------------------------------------
// 2) tn = ||I+A+S||_F^2 per (node, channel) + LayerNorm over channels.
//    One block (128 threads) per node. Output stored f16 (WMMA A operand).
// ---------------------------------------------------------------------------
__global__ void tn_layernorm_kernel(const float* __restrict__ acc,
                                    const float* __restrict__ ln_g,
                                    const float* __restrict__ ln_b,
                                    _Float16* __restrict__ tnorm16) // [N,128]
{
    const int n = blockIdx.x;
    const int h = threadIdx.x;
    const float* p = acc + ((size_t)n * HIDDEN + h) * 10;
    const float i  = p[0];
    const float ax = p[1], ay = p[2], az = p[3];
    const float sxx = p[4], syy = p[5], szz = p[6];
    const float sxy = p[7], sxz = p[8], syz = p[9];
    const float tn = 3.0f * i * i
                   + 2.0f * (ax * ax + ay * ay + az * az)
                   + sxx * sxx + syy * syy + szz * szz
                   + 2.0f * (sxy * sxy + sxz * sxz + syz * syz);

    __shared__ float s1[HIDDEN], s2[HIDDEN];
    s1[h] = tn; s2[h] = tn * tn;
    __syncthreads();
    for (int off = 64; off > 0; off >>= 1) {
        if (h < off) { s1[h] += s1[h + off]; s2[h] += s2[h + off]; }
        __syncthreads();
    }
    const float mean = s1[0] * (1.0f / HIDDEN);
    const float var  = s2[0] * (1.0f / HIDDEN) - mean * mean;
    const float inv  = rsqrtf(var + LN_EPS);
    tnorm16[(size_t)n * HIDDEN + h] =
        (_Float16)((tn - mean) * inv * ln_g[h] + ln_b[h]);
}

// ---------------------------------------------------------------------------
// 3) f16-WMMA GEMM: C[M,Nout] = silu(A[M,K] @ W[Nout,K]^T + bias)
//    A and W pre-staged f16; output f16 or f32 (template).
//    One wave per 16x16 output tile, 8 waves / block.
// ---------------------------------------------------------------------------
template <bool STORE_HALF>
__global__ void __launch_bounds__(256, 1)
gemm_bias_silu_kernel(const _Float16* __restrict__ A,
                      const _Float16* __restrict__ W,
                      const float* __restrict__ bias,
                      void* __restrict__ Cout,
                      int M, int Nout, int K)
{
    const int lane = threadIdx.x & 31;
    const int wave = threadIdx.x >> 5;
    const int tilesM = (M + 15) >> 4;
    const int tilesN = Nout >> 4;
    const int tid = blockIdx.x * 8 + wave;
    if (tid >= tilesM * tilesN) return;          // wave-uniform
    const int tm = tid % tilesM;
    const int tn = tid / tilesM;

    int rowA = tm * 16 + (lane & 15);
    if (rowA >= M) rowA = M - 1;                 // clamp loads, guard stores
    const int col = tn * 16 + (lane & 15);

    v8f accv = {};
#pragma unroll 1
    for (int k0 = 0; k0 < K; k0 += 32) {
        v16h a, b;
#pragma unroll
        for (int j = 0; j < 16; ++j) {
            const int k = k0 + kmap(lane, j);
            a[j] = A[(size_t)rowA * K + k];
            b[j] = W[(size_t)col  * K + k];
        }
        accv = wmma_f16(a, b, accv);
    }
    const float bv = bias[col];
#pragma unroll
    for (int r = 0; r < 8; ++r) {
        const int row = tm * 16 + ((lane >> 4) << 3) + r;
        if (row < M) {
            float x = accv[r] + bv;
            x = x / (1.0f + __expf(-x));         // silu
            if (STORE_HALF)
                ((_Float16*)Cout)[(size_t)row * Nout + col] = (_Float16)x;
            else
                ((float*)Cout)[(size_t)row * Nout + col] = x;
        }
    }
}

// ---------------------------------------------------------------------------
// 4) Channel mixing (Wt1/Wt2/Wt3, pre-staged f16) on the 10-component
//    representation, fused with norm3 scaling and 3x3 output assembly.
//    One wave per (16-node, 16-channel) tile; 10 WMMA accumulations.
// ---------------------------------------------------------------------------
__global__ void __launch_bounds__(256, 1)
mix_assemble_kernel(const float*    __restrict__ acc,   // [N,128,10] f32
                    const _Float16* __restrict__ Wt1_16,
                    const _Float16* __restrict__ Wt2_16,
                    const _Float16* __restrict__ Wt3_16,
                    const float*    __restrict__ y2,    // [N,384] f32
                    float* __restrict__ out)            // [N,128,3,3]
{
    const int lane = threadIdx.x & 31;
    const int wave = threadIdx.x >> 5;
    const int tilesM = (N_NODES + 15) >> 4;      // 313
    const int tid = blockIdx.x * 8 + wave;
    if (tid >= tilesM * (HIDDEN / 16)) return;   // wave-uniform
    const int tm = tid % tilesM;
    const int gc = tid / tilesM;

    int rowA = tm * 16 + (lane & 15);
    if (rowA >= N_NODES) rowA = N_NODES - 1;
    const int g = gc * 16 + (lane & 15);

    v8f res[10];
#pragma unroll
    for (int c = 0; c < 10; ++c) {
        const _Float16* Wt = (c == 0) ? Wt1_16 : (c < 4) ? Wt2_16 : Wt3_16;
        v8f a_ = {};
#pragma unroll
        for (int k0 = 0; k0 < HIDDEN; k0 += 32) {
            v16h a, b;
#pragma unroll
            for (int j = 0; j < 16; ++j) {
                const int hh = k0 + kmap(lane, j);
                a[j] = (_Float16)acc[((size_t)rowA * HIDDEN + hh) * 10 + c];
                b[j] = Wt[(size_t)g * HIDDEN + hh];
            }
            a_ = wmma_f16(a, b, a_);
        }
        res[c] = a_;
    }

#pragma unroll
    for (int r = 0; r < 8; ++r) {
        const int node = tm * 16 + ((lane >> 4) << 3) + r;
        if (node >= N_NODES) continue;
        const float* nm = y2 + (size_t)node * 384 + g * 3;
        const float n0 = nm[0], n1 = nm[1], n2 = nm[2];
        const float ip  = res[0][r] * n0;
        const float axp = res[1][r] * n1;
        const float ayp = res[2][r] * n1;
        const float azp = res[3][r] * n1;
        const float sxxp = res[4][r] * n2;
        const float syyp = res[5][r] * n2;
        const float szzp = res[6][r] * n2;
        const float sxyp = res[7][r] * n2;
        const float sxzp = res[8][r] * n2;
        const float syzp = res[9][r] * n2;
        float* o = out + ((size_t)node * HIDDEN + g) * 9;
        o[0] = ip + sxxp;          // [0,0]
        o[1] = -azp + sxyp;        // [0,1]
        o[2] =  ayp + sxzp;        // [0,2]
        o[3] =  azp + sxyp;        // [1,0]
        o[4] = ip + syyp;          // [1,1]
        o[5] = -axp + syzp;        // [1,2]
        o[6] = -ayp + sxzp;        // [2,0]
        o[7] =  axp + syzp;        // [2,1]
        o[8] = ip + szzp;          // [2,2]
    }
}

// ---------------------------------------------------------------------------
// Launch
// ---------------------------------------------------------------------------
extern "C" void kernel_launch(void* const* d_in, const int* in_sizes, int n_in,
                              void* d_out, int out_size, void* d_ws, size_t ws_size,
                              hipStream_t stream) {
    const int*   z        = (const int*)  d_in[0];
    const int*   eidx     = (const int*)  d_in[1];
    const float* ew       = (const float*)d_in[2];
    const float* evec     = (const float*)d_in[3];
    const float* eattr    = (const float*)d_in[4];
    const float* emb      = (const float*)d_in[5];
    const float* W_emb2   = (const float*)d_in[6];
    const float* b_emb2   = (const float*)d_in[7];
    const float* Wd1      = (const float*)d_in[8];
    const float* bd1      = (const float*)d_in[9];
    const float* Wd2      = (const float*)d_in[10];
    const float* bd2      = (const float*)d_in[11];
    const float* Wd3      = (const float*)d_in[12];
    const float* bd3      = (const float*)d_in[13];
    const float* Wt1      = (const float*)d_in[14];
    const float* Wt2      = (const float*)d_in[15];
    const float* Wt3      = (const float*)d_in[16];
    const float* Ws1      = (const float*)d_in[17];
    const float* bs1      = (const float*)d_in[18];
    const float* Ws2      = (const float*)d_in[19];
    const float* bs2      = (const float*)d_in[20];
    const float* ln_g     = (const float*)d_in[21];
    const float* ln_b     = (const float*)d_in[22];
    float* out = (float*)d_out;

    // ---- workspace layout ----
    // f32 region: acc [5000*128*10] | y2 [5000*384]
    float* acc = (float*)d_ws;
    float* y2  = acc + (size_t)N_NODES * HIDDEN * 10;
    // f16 region after the f32 region
    _Float16* h16 = (_Float16*)(y2 + (size_t)N_NODES * 384);
    _Float16* tnorm16 = h16;                 h16 += (size_t)N_NODES * HIDDEN;
    _Float16* y1_16   = h16;                 h16 += (size_t)N_NODES * 256;
    _Float16* emb16   = h16;                 h16 += 128 * HIDDEN;
    // contiguous weight block staged into LDS by the edge kernel:
    _Float16* wblk16  = h16;                 // Wemb2 | Wd1 | Wd2 | Wd3
    _Float16* Wemb2_16= h16;                 h16 += HIDDEN * 256;
    _Float16* Wd1_16  = h16;                 h16 += HIDDEN * NUM_RBF;
    _Float16* Wd2_16  = h16;                 h16 += HIDDEN * NUM_RBF;
    _Float16* Wd3_16  = h16;                 h16 += HIDDEN * NUM_RBF;
    _Float16* Wt1_16  = h16;                 h16 += HIDDEN * HIDDEN;
    _Float16* Wt2_16  = h16;                 h16 += HIDDEN * HIDDEN;
    _Float16* Wt3_16  = h16;                 h16 += HIDDEN * HIDDEN;
    _Float16* Ws1_16  = h16;                 h16 += 256 * HIDDEN;
    _Float16* Ws2_16  = h16;                 h16 += 384 * 256;

    // ---- stage weights to f16 (all stationary operands) ----
    cvt_f16_kernel<<<64, 256, 0, stream>>>(emb,    emb16,    128 * HIDDEN);
    cvt_f16_kernel<<<64, 256, 0, stream>>>(W_emb2, Wemb2_16, HIDDEN * 256);
    cvt_f16_kernel<<<64, 256, 0, stream>>>(Wd1,    Wd1_16,   HIDDEN * NUM_RBF);
    cvt_f16_kernel<<<64, 256, 0, stream>>>(Wd2,    Wd2_16,   HIDDEN * NUM_RBF);
    cvt_f16_kernel<<<64, 256, 0, stream>>>(Wd3,    Wd3_16,   HIDDEN * NUM_RBF);
    cvt_f16_kernel<<<64, 256, 0, stream>>>(Wt1,    Wt1_16,   HIDDEN * HIDDEN);
    cvt_f16_kernel<<<64, 256, 0, stream>>>(Wt2,    Wt2_16,   HIDDEN * HIDDEN);
    cvt_f16_kernel<<<64, 256, 0, stream>>>(Wt3,    Wt3_16,   HIDDEN * HIDDEN);
    cvt_f16_kernel<<<64, 256, 0, stream>>>(Ws1,    Ws1_16,   256 * HIDDEN);
    cvt_f16_kernel<<<256, 256, 0, stream>>>(Ws2,   Ws2_16,   384 * 256);

    const int accN = N_NODES * HIDDEN * 10;
    zero_f32_kernel<<<2048, 256, 0, stream>>>(acc, accN);

    // 1) fused edge GEMMs + scatter (5000 tiles, 8 waves/block, 112KB LDS)
    edge_kernel<<<(N_EDGES / 16 + 7) / 8, 256, LDS_W_BYTES, stream>>>(
        z, eidx, ew, evec, eattr, emb16, wblk16, b_emb2,
        bd1, bd2, bd3, acc);

    // 2) tn + LayerNorm (f16 output: next consumer is a WMMA A operand)
    tn_layernorm_kernel<<<N_NODES, HIDDEN, 0, stream>>>(acc, ln_g, ln_b, tnorm16);

    // 3) scalar MLP: [N,128]->[N,256] (f16) ->[N,384] (f32), both silu
    {
        const int tilesM = (N_NODES + 15) >> 4;
        const int t1 = tilesM * (256 / 16);
        gemm_bias_silu_kernel<true><<<(t1 + 7) / 8, 256, 0, stream>>>(
            tnorm16, Ws1_16, bs1, (void*)y1_16, N_NODES, 256, HIDDEN);
        const int t2 = tilesM * (384 / 16);
        gemm_bias_silu_kernel<false><<<(t2 + 7) / 8, 256, 0, stream>>>(
            y1_16, Ws2_16, bs2, (void*)y2, N_NODES, 384, 256);
    }

    // 4) channel mixing + norm3 scaling + 3x3 assembly
    {
        const int tilesM = (N_NODES + 15) >> 4;
        const int t = tilesM * (HIDDEN / 16);
        mix_assemble_kernel<<<(t + 7) / 8, 256, 0, stream>>>(
            acc, Wt1_16, Wt2_16, Wt3_16, y2, out);
    }
}
